// ProsMMA_68891275427851
// MI455X (gfx1250) — matrix operand order repaired and checked by reading, not compiled, not run on hardware
//
#include <hip/hip_runtime.h>
#include <hip/hip_bf16.h>
#include <cmath>

// ---------------------------------------------------------------------------
// MI455X (gfx1250, wave32) implementation.
// Dense contractions: v_wmma_f32_16x16x32_f16 (f16 in, f32 acc).
// Tile staging: global_load_async_to_lds_b128 (ASYNCcnt) where rows are
// contiguous + aligned, scalar zero-fill fallback for ragged tiles.
// ---------------------------------------------------------------------------

typedef __attribute__((ext_vector_type(16))) _Float16 v16h;
typedef __attribute__((ext_vector_type(8)))  float    v8f;

#define DEVI __device__ __forceinline__

DEVI v8f wmma_f16(v16h a, v16h b, v8f c) {
  // D = A(16x32) * B(32x16) + C(16x16), f32 accumulate
  return __builtin_amdgcn_wmma_f32_16x16x32_f16(false, a, false, b, (short)0, c,
                                                false, false);
}

// Async DMA: 16 bytes global -> LDS, tracked by ASYNCcnt (no VGPR data path).
// GVS addressing: mem = SADDR + VADDR32; VDST = LDS byte address.
DEVI void async_load_b128(unsigned lds_addr, unsigned voff, const void* sbase) {
  asm volatile("global_load_async_to_lds_b128 %0, %1, %2"
               :: "v"(lds_addr), "v"(voff), "s"(sbase)
               : "memory");
}
DEVI void wait_async0() { asm volatile("s_wait_asynccnt 0x0" ::: "memory"); }

// A fragment: MxK (K=32 slice) from row-major LDS tile (stride `ld` halfs).
// Lane L holds row m = row0 + (L&15); K interleave per ISA 16-bit A layout.
DEVI v16h frag_a(const _Float16* tile, int row0, int ld, int k0, int lane) {
  const _Float16* r = tile + (size_t)(row0 + (lane & 15)) * ld + k0;
  const int g = (lane >> 4) * 8;
  v16h a;
#pragma unroll
  for (int e = 0; e < 8; ++e) { a[e] = r[g + e]; a[8 + e] = r[16 + g + e]; }
  return a;
}

// B fragment: KxN (K=32 slice). LDS tile stored transposed Bt[n][k] (stride ld).
DEVI v16h frag_b(const _Float16* tileT, int col0, int ld, int k0, int lane) {
  const _Float16* r = tileT + (size_t)(col0 + (lane & 15)) * ld + k0;
  const int g = (lane >> 4) * 8;
  v16h b;
#pragma unroll
  for (int e = 0; e < 8; ++e) { b[e] = r[g + e]; b[8 + e] = r[16 + g + e]; }
  return b;
}

DEVI float gelu_erf(float x) { return 0.5f * x * (1.0f + erff(x * 0.70710678118654752f)); }
// Fast sigmoid: v_exp_f32 + v_rcp_f32 (avoid IEEE div_scale/Newton chains).
DEVI float sigmoid_f(float x) { return __builtin_amdgcn_rcpf(1.0f + __expf(-x)); }

enum {
  EPI_BIAS_M = 1,   // bias indexed by output row m
  EPI_BIAS_N = 2,   // bias indexed by output col n
  EPI_GELU   = 4,
  EPI_RES    = 8,   // add f32 residual (ldc layout)
  EPI_F16    = 16,  // write f16 copy
  EPI_F32    = 32,  // write f32 result
  EPI_ATOMIC = 64   // atomicAdd f32 result (split-K)
};

// Generic batched GEMM: C[m,n] = sum_k A[m,k]*B[k,n], 64x64 tile, 8 waves.
template <int EPI, bool SPLITK>
__global__ __launch_bounds__(256) void gemm_ws(
    const _Float16* __restrict__ A, long long strideA, int lda,
    const _Float16* __restrict__ Bm, long long strideB, int ldb,
    float* __restrict__ Cf, long long strideC, int ldc,
    _Float16* __restrict__ Oh, long long strideOh,
    const float* __restrict__ bias,
    const float* __restrict__ Res, long long strideRes,
    int M, int N, int K, int kChunk) {
  __shared__ __align__(16) _Float16 sA[64 * 32];   // row-major M x K slice
  __shared__ __align__(16) _Float16 sBt[64 * 32];  // transposed: Bt[n][k]

  const int t = threadIdx.x;
  const int lane = t & 31;
  const int wv = t >> 5;              // 8 wave32 waves
  const int m0 = blockIdx.x * 64;
  const int n0 = blockIdx.y * 64;
  const int bz = SPLITK ? 0 : blockIdx.z;
  const int kBeg = SPLITK ? blockIdx.z * kChunk : 0;
  int kEnd = SPLITK ? (kBeg + kChunk) : K;
  if (kEnd > K) kEnd = K;

  const _Float16* Ab = A + (long long)bz * strideA;
  const _Float16* Bb = Bm + (long long)bz * strideB;

  const unsigned ldsA0 = (unsigned)(size_t)&sA[0];
  // Async path legal when every 16B chunk is in-bounds and 16B aligned.
  const bool alignedA =
      ((lda & 7) == 0) && ((((size_t)Ab) & 15) == 0) && (m0 + 64 <= M);

  const int mi = wv & 3;
  const int njb = (wv >> 2) * 2;

  v8f acc0 = {0.f, 0.f, 0.f, 0.f, 0.f, 0.f, 0.f, 0.f};
  v8f acc1 = acc0;

  for (int k0 = kBeg; k0 < kEnd; k0 += 32) {
    const bool aAsync = alignedA && (k0 + 32 <= kEnd);  // block-uniform
    if (aAsync) {
      // 256 lanes x 16B = full 64x32 f16 A tile, one DMA op per lane.
      int gm = m0 + (t >> 2);
      unsigned gk = (unsigned)k0 + (unsigned)(t & 3) * 8u;
      unsigned voff = (unsigned)(((size_t)gm * (size_t)lda + gk) * 2);
      async_load_b128(ldsA0 + (unsigned)t * 16u, voff, (const void*)Ab);
    } else {
#pragma unroll
      for (int i = 0; i < 8; ++i) {
        int lin = t * 8 + i;
        int r = lin >> 5, c = lin & 31;
        int gm = m0 + r, gk = k0 + c;
        sA[lin] =
            (gm < M && gk < kEnd) ? Ab[(size_t)gm * lda + gk] : (_Float16)0.f;
      }
    }
#pragma unroll
    for (int i = 0; i < 8; ++i) {  // B stage is a transpose -> scalar gather
      int lin = t * 8 + i;
      int r = lin >> 5, c = lin & 31;
      int gn = n0 + r, gk = k0 + c;
      sBt[lin] =
          (gn < N && gk < kEnd) ? Bb[(size_t)gk * ldb + gn] : (_Float16)0.f;
    }
    if (k0 + 32 < kEnd) {  // speculative prefetch of next A tile rows
      int pm = m0 + (t & 63);
      if (pm >= M) pm = M - 1;
      __builtin_prefetch(&Ab[(size_t)pm * lda + (k0 + 32)], 0, 1);
    }
    if (aAsync) wait_async0();
    __syncthreads();

    v16h af = frag_a(sA, mi * 16, 32, 0, lane);
    v16h bf0 = frag_b(sBt, njb * 16, 32, 0, lane);
    v16h bf1 = frag_b(sBt, (njb + 1) * 16, 32, 0, lane);
    acc0 = wmma_f16(af, bf0, acc0);
    acc1 = wmma_f16(af, bf1, acc1);
    __syncthreads();
  }

#pragma unroll
  for (int j = 0; j < 2; ++j) {
    v8f& acc = j ? acc1 : acc0;
    int n = n0 + (njb + j) * 16 + (lane & 15);
    if (n >= N) continue;
#pragma unroll
    for (int r = 0; r < 8; ++r) {
      int m = m0 + mi * 16 + r + 8 * (lane >> 4);
      if (m >= M) continue;
      float v = acc[r];
      if (EPI & EPI_BIAS_M) v += bias[m];
      if (EPI & EPI_BIAS_N) v += bias[n];
      if (EPI & EPI_GELU) v = gelu_erf(v);
      if (EPI & EPI_RES) v += Res[(long long)bz * strideRes + (size_t)m * ldc + n];
      if (EPI & EPI_ATOMIC) {
        atomicAdd(&Cf[(size_t)m * ldc + n], v);
      } else if (EPI & EPI_F32) {
        Cf[(long long)bz * strideC + (size_t)m * ldc + n] = v;
      }
      if (EPI & EPI_F16)
        Oh[(long long)bz * strideOh + (size_t)m * ldc + n] = (_Float16)v;
    }
  }
}

// ---------------------------------------------------------------------------
// Conv1d (NCH, OIH) + bias -> Y (B, Lout, C=256) f32.  IN_CH=64, LEN=512.
// ---------------------------------------------------------------------------
template <int KS, int STR, int PAD>
__global__ __launch_bounds__(256) void conv1d_ws(
    const float* __restrict__ X, const float* __restrict__ W,
    const float* __restrict__ bias, float* __restrict__ Y, int Lout) {
  int idx = blockIdx.x * 256 + threadIdx.x;
  if (idx >= 32 * Lout * 256) return;
  int c = idx & 255;
  int bl = idx >> 8;
  int l = bl % Lout;
  int b = bl / Lout;
  float acc = bias[c];
#pragma unroll
  for (int tt = 0; tt < KS; ++tt) {
    int pos = l * STR - PAD + tt;
    if (pos < 0 || pos >= 512) continue;
    const float* xp = X + (size_t)b * 64 * 512 + pos;
    const float* wp = W + (size_t)c * 64 * KS + tt;
#pragma unroll 8
    for (int ic = 0; ic < 64; ++ic) acc += xp[ic * 512] * wp[ic * KS];
  }
  Y[idx] = acc;
}

// ---------------------------------------------------------------------------
// LayerNorm over C=256 (one row per block). Optional 0.5*(x+x2) pre-combine.
// ---------------------------------------------------------------------------
__global__ __launch_bounds__(256) void ln256_ws(
    const float* __restrict__ Xin, const float* __restrict__ Xin2,
    const float* __restrict__ g, const float* __restrict__ bta,
    _Float16* __restrict__ outH, float* __restrict__ outF) {
  __shared__ float red[256];
  const int row = blockIdx.x, t = threadIdx.x;
  const size_t off = (size_t)row * 256 + t;
  float x = Xin[off];
  if (Xin2) x = 0.5f * (x + Xin2[off]);
  red[t] = x;
  __syncthreads();
  for (int s = 128; s > 0; s >>= 1) {
    if (t < s) red[t] += red[t + s];
    __syncthreads();
  }
  float mean = red[0] * (1.0f / 256.0f);
  __syncthreads();
  float d = x - mean;
  red[t] = d * d;
  __syncthreads();
  for (int s = 128; s > 0; s >>= 1) {
    if (t < s) red[t] += red[t + s];
    __syncthreads();
  }
  float var = red[0] * (1.0f / 256.0f);
  float o = d * rsqrtf(var + 1e-5f) * g[t] + bta[t];
  if (outH) outH[off] = (_Float16)o;
  if (outF) outF[off] = o;
}

// ---------------------------------------------------------------------------
// DiffAttention core. For one seq position s and a 64-row query tile over the
// n = b*8+h axis (N=256): scores = sigmoid(Q.K^T), normalize over keys l,
// out = attn @ V. Q/K/V/O are (B, S, 256) f16, head-packed channels.
// Grid: (S, 4). Dynamic LDS ~68 KB (WGP has 320 KB).
// ---------------------------------------------------------------------------
__global__ __launch_bounds__(256) void attn_ws(
    const _Float16* __restrict__ Qh, const _Float16* __restrict__ Kh,
    const _Float16* __restrict__ Vh, _Float16* __restrict__ Oh, int S) {
  extern __shared__ char smem[];
  _Float16* sQ = (_Float16*)smem;         // 64 x 32
  _Float16* sK = sQ + 64 * 32;            // 256 x 32   (Bt layout: [l][m])
  _Float16* sVt = sK + 256 * 32;          // 32 x 256   (Bt layout: [d][l])
  _Float16* sAttn = sVt + 32 * 256;       // 64 x 256
  float* sRsum = (float*)(sAttn + 64 * 256);  // 64

  const int t = threadIdx.x, lane = t & 31, wv = t >> 5;
  const int s = blockIdx.x;
  const int n0 = blockIdx.y * 64;

  if (t < 64) sRsum[t] = 0.f;

  // Q tile via async DMA: 128 x 16B chunks (rows are contiguous, 16B aligned).
  const unsigned ldsQ0 = (unsigned)(size_t)sQ;
  if (t < 128) {
    int r = t >> 2, m8 = (t & 3) * 8;
    int n = n0 + r;
    unsigned voff =
        (unsigned)((((size_t)(n >> 3) * S + s) * 256 + (n & 7) * 32 + m8) * 2);
    async_load_b128(ldsQ0 + (unsigned)t * 16u, voff, (const void*)Qh);
  }
  // K (all 256 keys) via async DMA: 1024 x 16B chunks, 4 per lane.
  const unsigned ldsK0 = (unsigned)(size_t)sK;
#pragma unroll
  for (int i = 0; i < 4; ++i) {
    int q = t * 4 + i;
    int l = q >> 2, m8 = (q & 3) * 8;
    unsigned voff =
        (unsigned)((((size_t)(l >> 3) * S + s) * 256 + (l & 7) * 32 + m8) * 2);
    async_load_b128(ldsK0 + (unsigned)q * 16u, voff, (const void*)Kh);
  }
#pragma unroll
  for (int i = 0; i < 32; ++i) {  // V transposed [d][l]: strided gather, scalar
    int lin = t * 32 + i;
    int d = lin >> 8, l = lin & 255;
    sVt[lin] = Vh[(((size_t)(l >> 3)) * S + s) * 256 + (l & 7) * 32 + d];
  }
  wait_async0();
  __syncthreads();

  {  // GEMM1: 64x256 scores, K=32; sigmoid + register row-sum accumulation
    const int mi = wv & 3;
    v16h af = frag_a(sQ, mi * 16, 32, 0, lane);
    float rowAcc[8] = {0.f, 0.f, 0.f, 0.f, 0.f, 0.f, 0.f, 0.f};
#pragma unroll
    for (int jj = 0; jj < 8; ++jj) {
      int lj = (wv >> 2) * 8 + jj;
      v16h bf = frag_b(sK, lj * 16, 32, 0, lane);
      v8f z = {0.f, 0.f, 0.f, 0.f, 0.f, 0.f, 0.f, 0.f};
      v8f c = wmma_f16(af, bf, z);
      int l = lj * 16 + (lane & 15);
#pragma unroll
      for (int r = 0; r < 8; ++r) {
        int m = mi * 16 + r + 8 * (lane >> 4);
        float sg = sigmoid_f(c[r]);
        sAttn[(size_t)m * 256 + l] = (_Float16)sg;
        rowAcc[r] += sg;
      }
    }
#pragma unroll
    for (int r = 0; r < 8; ++r)  // one ds_add_f32 per accumulator row
      atomicAdd(&sRsum[mi * 16 + r + 8 * (lane >> 4)], rowAcc[r]);
  }
  __syncthreads();
  for (int i = t; i < 64 * 256; i += 256) {  // normalize over keys (v_rcp_f32)
    int m = i >> 8;
    float rinv = __builtin_amdgcn_rcpf(sRsum[m]);
    sAttn[i] = (_Float16)((float)sAttn[i] * rinv);
  }
  __syncthreads();

  {  // GEMM2: out 64x32 = attn(64x256) @ V(256x32)
    const int mi = wv >> 1, dj = wv & 1;
    v8f acc = {0.f, 0.f, 0.f, 0.f, 0.f, 0.f, 0.f, 0.f};
#pragma unroll
    for (int kt = 0; kt < 8; ++kt) {
      v16h af = frag_a(sAttn, mi * 16, 256, kt * 32, lane);
      v16h bf = frag_b(sVt, dj * 16, 256, kt * 32, lane);
      acc = wmma_f16(af, bf, acc);
    }
    int d = dj * 16 + (lane & 15);
#pragma unroll
    for (int r = 0; r < 8; ++r) {
      int n = n0 + mi * 16 + r + 8 * (lane >> 4);
      Oh[(((size_t)(n >> 3)) * S + s) * 256 + (n & 7) * 32 + d] = (_Float16)acc[r];
    }
  }
}

// ---------------------------------------------------------------------------
// Small utility kernels.
// ---------------------------------------------------------------------------
__global__ __launch_bounds__(256) void cvt_f32_f16(const float* __restrict__ in,
                                                   _Float16* __restrict__ out,
                                                   long long n) {
  long long i = (long long)blockIdx.x * 256 + threadIdx.x;
  if (i < n) out[i] = (_Float16)in[i];
}

// in: (I x J) f32 row-major  ->  out: (J x I) f16 row-major
__global__ __launch_bounds__(256) void tcvt_f32_f16(const float* __restrict__ in,
                                                    _Float16* __restrict__ out,
                                                    int I, int J) {
  int idx = blockIdx.x * 256 + threadIdx.x;
  if (idx >= I * J) return;
  int i = idx / J, j = idx % J;
  out[(size_t)j * I + i] = (_Float16)in[idx];
}

__global__ __launch_bounds__(256) void init_acc_ws(const float* b0, const float* b1,
                                                   const float* b2, float* acc) {
  int i = blockIdx.x * 256 + threadIdx.x;
  if (i < 32 * 128) acc[i] = b0[i & 127] + b1[i & 127] + b2[i & 127];
}

__global__ __launch_bounds__(64) void fc_ws(const float* __restrict__ acc,
                                            const float* __restrict__ fw,
                                            const float* __restrict__ fb,
                                            float* __restrict__ out) {
  int t = threadIdx.x;
  if (t >= 64) return;
  int b = t >> 1, o = t & 1;
  float s = fb[o];
#pragma unroll 16
  for (int j = 0; j < 128; ++j) s += acc[b * 128 + j] * fw[j * 2 + o];
  out[b * 2 + o] = s;
}

// ---------------------------------------------------------------------------
// Host orchestration.
// ---------------------------------------------------------------------------
static inline int cdiv(long long a, long long b) { return (int)((a + b - 1) / b); }

#define GEMM(EPI_, SK_, A_, sA_, lda_, B_, sB_, ldb_, C_, sC_, ldc_, OH_, sOH_, \
             bias_, Res_, sRes_, M_, N_, K_, kC_, gz_)                          \
  gemm_ws<(EPI_), SK_><<<dim3(cdiv((M_), 64), cdiv((N_), 64), (gz_)), 256, 0,   \
                         stream>>>((A_), (sA_), (lda_), (B_), (sB_), (ldb_),    \
                                   (C_), (sC_), (ldc_), (OH_), (sOH_), (bias_), \
                                   (Res_), (sRes_), (M_), (N_), (K_), (kC_))

extern "C" void kernel_launch(void* const* d_in, const int* in_sizes, int n_in,
                              void* d_out, int out_size, void* d_ws,
                              size_t ws_size, hipStream_t stream) {
  (void)in_sizes; (void)n_in; (void)out_size; (void)ws_size;
  auto F = [&](int i) { return (const float*)d_in[i]; };
  int ix = 0;
  // Input order: jax tree flatten (dicts sorted by key, lists in order).
  // {'X','params'} -> X first; params: branches, fc_b, fc_w.
  const float* X = F(ix++);
  struct BrP {
    const float *cb1, *cb2, *conv_b, *conv_w, *cw1, *cw2;
    const float *Wk[2], *Wo[2], *Wq[2], *Wv[2], *ln_b[2], *ln_g[2];
    const float *lin_b, *lin_w, *ln1_b, *ln1_g, *ln2_b, *ln2_g;
    const float *tb1, *tb2, *tw1, *tw2;
  } P[3];
  for (int br = 0; br < 3; ++br) {
    BrP& p = P[br];
    p.cb1 = F(ix++); p.cb2 = F(ix++); p.conv_b = F(ix++); p.conv_w = F(ix++);
    p.cw1 = F(ix++); p.cw2 = F(ix++);
    for (int l = 0; l < 2; ++l) {  // sorted: Wk, Wo, Wq, Wv, ln_b, ln_g
      p.Wk[l] = F(ix++); p.Wo[l] = F(ix++); p.Wq[l] = F(ix++); p.Wv[l] = F(ix++);
      p.ln_b[l] = F(ix++); p.ln_g[l] = F(ix++);
    }
    p.lin_b = F(ix++); p.lin_w = F(ix++);
    p.ln1_b = F(ix++); p.ln1_g = F(ix++); p.ln2_b = F(ix++); p.ln2_g = F(ix++);
    p.tb1 = F(ix++); p.tb2 = F(ix++); p.tw1 = F(ix++); p.tw2 = F(ix++);
  }
  const float* fc_b = F(ix++);
  const float* fc_w = F(ix++);

  // Workspace arena (sized for branch 0, reused sequentially per branch).
  char* ws = (char*)d_ws;
  const size_t O_Y   = 0;                       // y   f32 (B,L,256)   16.8 MB
  const size_t O_YH  = O_Y + 16777216;          // yh  f16              8.4 MB
  const size_t O_LN  = O_YH + 8388608;          // LN out f16           8.4 MB
  const size_t O_T   = O_LN + 8388608;          // mixer temp f16      16.8 MB
  const size_t O_Q   = O_T + 16777216;          // Qh f16               8.4 MB
  const size_t O_K   = O_Q + 8388608;           // Kh f16               8.4 MB
  const size_t O_V   = O_K + 8388608;           // Vh f16               8.4 MB
  const size_t O_O   = O_V + 8388608;           // Oh f16               8.4 MB
  const size_t O_P   = O_O + 8388608;           // P  f32              16.8 MB
  const size_t O_W   = O_P + 16777216;          // f16 weight scratch  33.6 MB
  const size_t O_ACC = O_W + 33554432;          // acc 32x128 f32
  float*    y   = (float*)(ws + O_Y);
  _Float16* yh  = (_Float16*)(ws + O_YH);
  _Float16* lnh = (_Float16*)(ws + O_LN);
  _Float16* Th  = (_Float16*)(ws + O_T);
  _Float16* Qh  = (_Float16*)(ws + O_Q);
  _Float16* Kh  = (_Float16*)(ws + O_K);
  _Float16* Vh  = (_Float16*)(ws + O_V);
  _Float16* Ohh = (_Float16*)(ws + O_O);
  float*    Pf  = (float*)(ws + O_P);
  _Float16* wh  = (_Float16*)(ws + O_W);
  float*    acc = (float*)(ws + O_ACC);

  const int Ls[3] = {512, 256, 171};
  const int ATTN_SMEM = (64 * 32 + 256 * 32 + 32 * 256 + 64 * 256) * 2 + 64 * 4;

  init_acc_ws<<<16, 256, 0, stream>>>(P[0].lin_b, P[1].lin_b, P[2].lin_b, acc);

  for (int br = 0; br < 3; ++br) {
    const BrP& p = P[br];
    const int L = Ls[br], TI = 2 * L;
    const long long LC = (long long)L * 256;
    const int RB = 32 * L;

    // 1. conv + bias -> y (B,L,256) f32
    int cg = cdiv((long long)32 * L * 256, 256);
    if (br == 0) conv1d_ws<1, 1, 0><<<cg, 256, 0, stream>>>(X, p.conv_w, p.conv_b, y, L);
    if (br == 1) conv1d_ws<3, 2, 1><<<cg, 256, 0, stream>>>(X, p.conv_w, p.conv_b, y, L);
    if (br == 2) conv1d_ws<5, 3, 2><<<cg, 256, 0, stream>>>(X, p.conv_w, p.conv_b, y, L);

    // 2. token mixing: T = GELU(tw1^T @ LN1(y) + tb1); y += tw2^T @ T + tb2
    ln256_ws<<<RB, 256, 0, stream>>>(y, nullptr, p.ln1_g, p.ln1_b, lnh, nullptr);
    tcvt_f32_f16<<<cdiv((long long)L * TI, 256), 256, 0, stream>>>(p.tw1, wh, L, TI);
    GEMM(EPI_BIAS_M | EPI_GELU | EPI_F16, false, wh, 0LL, L, lnh, LC, 256,
         (float*)nullptr, 0LL, 256, Th, (long long)TI * 256, p.tb1,
         (const float*)nullptr, 0LL, TI, 256, L, 32, 32);
    tcvt_f32_f16<<<cdiv((long long)TI * L, 256), 256, 0, stream>>>(p.tw2, wh, TI, L);
    GEMM(EPI_BIAS_M | EPI_RES | EPI_F32, false, wh, 0LL, TI, Th,
         (long long)TI * 256, 256, y, LC, 256, (_Float16*)nullptr, 0LL, p.tb2,
         y, LC, L, 256, TI, 32, 32);

    // 3. channel mixing: y += GELU(LN2(y) @ cw1 + cb1) @ cw2 + cb2 (emit yh f16)
    ln256_ws<<<RB, 256, 0, stream>>>(y, nullptr, p.ln2_g, p.ln2_b, lnh, nullptr);
    cvt_f32_f16<<<cdiv(256LL * 512, 256), 256, 0, stream>>>(p.cw1, wh, 256LL * 512);
    GEMM(EPI_BIAS_N | EPI_GELU | EPI_F16, false, lnh, LC, 256, wh, 0LL, 512,
         (float*)nullptr, 0LL, 512, Th, (long long)L * 512, p.cb1,
         (const float*)nullptr, 0LL, L, 512, 256, 32, 32);
    cvt_f32_f16<<<cdiv(512LL * 256, 256), 256, 0, stream>>>(p.cw2, wh, 512LL * 256);
    GEMM(EPI_BIAS_N | EPI_RES | EPI_F32 | EPI_F16, false, Th,
         (long long)L * 512, 512, wh, 0LL, 256, y, LC, 256, yh, LC, p.cb2, y,
         LC, L, 256, 512, 32, 32);

    // 4. encoder layers: y = LN(0.5*y + 0.5*MHA(y))
    for (int ly = 0; ly < 2; ++ly) {
      cvt_f32_f16<<<cdiv(65536LL, 256), 256, 0, stream>>>(p.Wq[ly], wh, 65536LL);
      GEMM(EPI_F16, false, yh, LC, 256, wh, 0LL, 256, (float*)nullptr, 0LL, 256,
           Qh, LC, (const float*)nullptr, (const float*)nullptr, 0LL, L, 256,
           256, 32, 32);
      cvt_f32_f16<<<cdiv(65536LL, 256), 256, 0, stream>>>(p.Wk[ly], wh, 65536LL);
      GEMM(EPI_F16, false, yh, LC, 256, wh, 0LL, 256, (float*)nullptr, 0LL, 256,
           Kh, LC, (const float*)nullptr, (const float*)nullptr, 0LL, L, 256,
           256, 32, 32);
      cvt_f32_f16<<<cdiv(65536LL, 256), 256, 0, stream>>>(p.Wv[ly], wh, 65536LL);
      GEMM(EPI_F16, false, yh, LC, 256, wh, 0LL, 256, (float*)nullptr, 0LL, 256,
           Vh, LC, (const float*)nullptr, (const float*)nullptr, 0LL, L, 256,
           256, 32, 32);

      attn_ws<<<dim3(L, 4), 256, ATTN_SMEM, stream>>>(Qh, Kh, Vh, Ohh, L);

      cvt_f32_f16<<<cdiv(65536LL, 256), 256, 0, stream>>>(p.Wo[ly], wh, 65536LL);
      GEMM(EPI_F32, false, Ohh, LC, 256, wh, 0LL, 256, Pf, LC, 256,
           (_Float16*)nullptr, 0LL, (const float*)nullptr, (const float*)nullptr,
           0LL, L, 256, 256, 32, 32);

      ln256_ws<<<RB, 256, 0, stream>>>(y, Pf, p.ln_g[ly], p.ln_b[ly], yh, y);
    }

    // 5. final linear, split-K atomic accumulate into acc (32 x 128)
    const long long KL = LC;  // L*256
    cvt_f32_f16<<<cdiv(KL * 128, 256), 256, 0, stream>>>(p.lin_w, wh, KL * 128);
    const int kChunk = 8192;
    const int nz = cdiv(KL, kChunk);
    GEMM(EPI_ATOMIC, true, yh, 0LL, (int)KL, wh, 0LL, 128, acc, 0LL, 128,
         (_Float16*)nullptr, 0LL, (const float*)nullptr, (const float*)nullptr,
         0LL, 32, 128, (int)KL, kChunk, nz);
  }

  // 6. head: out (32 x 2) = acc @ fc_w + fc_b
  fc_ws<<<1, 64, 0, stream>>>(acc, fc_w, fc_b, (float*)d_out);
}
